// VSGNet_3633542332445
// MI455X (gfx1250) — matrix operand order repaired and checked by reading, not compiled, not run on hardware
//
#include <hip/hip_runtime.h>
#include <cstdint>

// ---------------- problem constants ----------------
#define BB   64
#define PP   16
#define LL   16
#define DD   1024
#define SP   32
#define PROJ 512
#define ACT  29
#define CC   (BB*PP*LL)     // 16384
#define BP   (BB*PP)        // 1024
#define BL   (BB*LL)        // 1024

typedef _Float16 h16;
typedef __attribute__((ext_vector_type(16))) _Float16 v16h;
typedef __attribute__((ext_vector_type(8)))  float    v8f;

// =====================================================================
// Direct-from-global f16 WMMA GEMM (no LDS, no barriers).
//   A : MxK row-major f16 (lda)        -> A-fragments are 2x b128 per lane
//   BT: NxK row-major f16 (ldb) == B^T -> B-fragments are 2x b128 per lane
// One wave32 computes an (MT*16) x (NT*16) macro-tile: MT*NT v_wmma per
// 32-wide K-step against MT+NT fragment loads. M % (MT*16) == 0,
// N % (NT*16) == 0, K % 32 == 0 (all hold for this net's shapes).
// =====================================================================
template<int MT, int NT>
__global__ __launch_bounds__(32)
void gemm_wmma_direct(const h16* __restrict__ A, int lda,
                      const h16* __restrict__ BT, int ldb,
                      const float* __restrict__ bias,
                      float* __restrict__ Cf, h16* __restrict__ Ch, int ldc,
                      int K, int relu)
{
    const int lane = threadIdx.x;     // 0..31
    const int n16  = lane & 15;
    const int hi   = lane >> 4;
    const int tm   = blockIdx.y * (MT * 16);
    const int tn   = blockIdx.x * (NT * 16);

    v8f acc[MT][NT];
#pragma unroll
    for (int i = 0; i < MT; ++i)
#pragma unroll
        for (int j = 0; j < NT; ++j)
            acc[i][j] = (v8f){0.f, 0.f, 0.f, 0.f, 0.f, 0.f, 0.f, 0.f};

    union Frag { v16h v; uint4 q[2]; };

    for (int k0 = 0; k0 < K; k0 += 32) {
        Frag a[MT], b[NT];
        // A fragment: lane = row m, halves [hi*8 .. +7] and [16+hi*8 .. +7]
#pragma unroll
        for (int i = 0; i < MT; ++i) {
            const h16* ap = A + (size_t)(tm + i * 16 + n16) * lda + k0 + hi * 8;
            a[i].q[0] = *(const uint4*)(ap);
            a[i].q[1] = *(const uint4*)(ap + 16);
        }
        // B fragment: lane = col n, halves [hi*8 .. +7] and [16+hi*8 .. +7] of B^T row n
#pragma unroll
        for (int j = 0; j < NT; ++j) {
            const h16* bp = BT + (size_t)(tn + j * 16 + n16) * ldb + k0 + hi * 8;
            b[j].q[0] = *(const uint4*)(bp);
            b[j].q[1] = *(const uint4*)(bp + 16);
        }
        // prefetch next K-slice (gfx1250: global_prefetch_b8)
        if (k0 + 32 < K) {
            __builtin_prefetch((const void*)(A + (size_t)(tm + n16) * lda + k0 + 32), 0, 1);
            __builtin_prefetch((const void*)(BT + (size_t)(tn + n16) * ldb + k0 + 32), 0, 1);
        }
#pragma unroll
        for (int i = 0; i < MT; ++i)
#pragma unroll
            for (int j = 0; j < NT; ++j)
                acc[i][j] = __builtin_amdgcn_wmma_f32_16x16x32_f16(
                                false, a[i].v, false, b[j].v,
                                (short)0, acc[i][j], false, false);
    }

    // store: lane holds column n16; accumulator VGPR r -> row r + 8*hi
#pragma unroll
    for (int i = 0; i < MT; ++i) {
#pragma unroll
        for (int j = 0; j < NT; ++j) {
            const int col = tn + j * 16 + n16;
            float bv = bias ? bias[col] : 0.f;
#pragma unroll
            for (int r = 0; r < 8; ++r) {
                const int row = tm + i * 16 + r + 8 * hi;
                float v = acc[i][j][r] + bv;
                if (relu) v = v > 0.f ? v : 0.f;
                const size_t off = (size_t)row * ldc + col;
                if (Cf) Cf[off] = v;
                if (Ch) Ch[off] = (h16)v;
            }
        }
    }
}

// ---------------- elementwise helpers ----------------
__global__ void cvt_f2h(const float* __restrict__ x, h16* __restrict__ y, int n) {
    int i = blockIdx.x * blockDim.x + threadIdx.x;
    if (i < n) y[i] = (h16)x[i];
}

// W [K,N] f32 (row-major) -> BT [N,K] f16 (row-major transpose)
__global__ void transpose_f2h(const float* __restrict__ W, h16* __restrict__ BT,
                              int K, int N) {
    int i = blockIdx.x * blockDim.x + threadIdx.x;
    if (i >= K * N) return;
    int n = i / K, k = i - n * K;
    BT[i] = (h16)W[(size_t)k * N + n];
}

// W [K,29] f32 -> BT [32,K] f16 transposed+padded; optional col 29 = extra (Wip)
__global__ void pad29T_f2h(const float* __restrict__ W, const float* __restrict__ extra,
                           h16* __restrict__ BT, int K) {
    int i = blockIdx.x * blockDim.x + threadIdx.x;
    if (i >= 32 * K) return;
    int n = i / K, k = i - n * K;
    float v = 0.f;
    if (n < ACT)                v = W[(size_t)k * ACT + n];
    else if (n == ACT && extra) v = extra[k];
    BT[i] = (h16)v;
}

// f_vis = relu(vis_p[b,p] + vis_o[b,l] + vis_c[b] + bias); f_ref = f_vis * a_ho
__global__ void fuse_fref(const float* __restrict__ vis_p, const float* __restrict__ vis_o,
                          const float* __restrict__ vis_c, const float* __restrict__ Wvis_b,
                          const h16* __restrict__ a_ho, h16* __restrict__ f_ref) {
    int i = blockIdx.x * blockDim.x + threadIdx.x;
    if (i >= CC * PROJ) return;
    int c = i >> 9, j = i & (PROJ - 1);
    int b = c >> 8, p = (c >> 4) & 15, l = c & 15;
    float fv = vis_p[(size_t)((b << 4) + p) * PROJ + j]
             + vis_o[(size_t)((b << 4) + l) * PROJ + j]
             + vis_c[(size_t)b * PROJ + j] + Wvis_b[j];
    fv = fv > 0.f ? fv : 0.f;
    f_ref[i] = (h16)(fv * (float)a_ho[i]);
}

// adjacency = sigmoid(i_ho) over object slots 1..15
__global__ void make_adj(const float* __restrict__ p_ref_pad, const float* __restrict__ Wip_b,
                         float* __restrict__ adj) {
    int i = blockIdx.x * blockDim.x + threadIdx.x;
    if (i >= BB * PP * 15) return;
    int o = i % 15, p = (i / 15) & 15, b = i / (15 * PP);
    int c = ((b * PP + p) * LL) + (o + 1);
    float z = p_ref_pad[(size_t)c * 32 + ACT] + Wip_b[0];
    adj[i] = 1.f / (1.f + __expf(-z));
}

// people_r = people + adj @ oh   (per (b,p,d), O=15)
__global__ void refine_people(const float* __restrict__ people, const float* __restrict__ adj,
                              const float* __restrict__ oh_all, h16* __restrict__ people_r) {
    int i = blockIdx.x * blockDim.x + threadIdx.x;
    if (i >= BP * DD) return;
    int d = i & (DD - 1);
    int row = i >> 10;            // b*16 + p
    int b = row >> 4;
    float s = people[i];
#pragma unroll 5
    for (int o = 0; o < 15; ++o)
        s += adj[row * 15 + o] * oh_all[(size_t)((b << 4) + (o + 1)) * DD + d];
    people_r[i] = (h16)s;
}

// objects_full: slot0 passthrough; slots 1..15 get adj^T @ ho
__global__ void refine_objects(const float* __restrict__ objects, const float* __restrict__ adj,
                               const float* __restrict__ ho_all, h16* __restrict__ objects_full) {
    int i = blockIdx.x * blockDim.x + threadIdx.x;
    if (i >= BL * DD) return;
    int d = i & (DD - 1);
    int row = i >> 10;            // b*16 + l
    int b = row >> 4, l = row & 15;
    float s = objects[i];
    if (l >= 1) {
#pragma unroll 4
        for (int p = 0; p < 16; ++p)
            s += adj[((b << 4) + p) * 15 + (l - 1)] * ho_all[(size_t)((b << 4) + p) * DD + d];
    }
    objects_full[i] = (h16)s;
}

// assemble (i_ho, p_ref, p_att, p_graph) into d_out (flat, concatenated)
__global__ void pack_out(const float* __restrict__ p_ref_pad, const float* __restrict__ p_att_pad,
                         const float* __restrict__ g_p, const float* __restrict__ g_o,
                         const float* __restrict__ Wip_b, const float* __restrict__ Wref_b,
                         const float* __restrict__ Watt_b, const float* __restrict__ Wg_b,
                         float* __restrict__ out) {
    int c = blockIdx.x * blockDim.x + threadIdx.x;
    if (c >= CC) return;
    int b = c >> 8, p = (c >> 4) & 15, l = c & 15;
    out[c] = p_ref_pad[(size_t)c * 32 + ACT] + Wip_b[0];     // i_ho logit
    float* pr = out + CC;
    float* pa = out + CC + (size_t)CC * ACT;
    float* pg = out + CC + 2 * (size_t)CC * ACT;
    const int rp = (b << 4) + p, ro = (b << 4) + l;
#pragma unroll
    for (int j = 0; j < ACT; ++j) {
        pr[(size_t)c * ACT + j] = p_ref_pad[(size_t)c * 32 + j] + Wref_b[j];
        pa[(size_t)c * ACT + j] = p_att_pad[(size_t)c * 32 + j] + Watt_b[j];
        pg[(size_t)c * ACT + j] = g_p[(size_t)rp * 32 + j] + g_o[(size_t)ro * 32 + j] + Wg_b[j];
    }
}

// =====================================================================
extern "C" void kernel_launch(void* const* d_in, const int* in_sizes, int n_in,
                              void* d_out, int out_size, void* d_ws, size_t ws_size,
                              hipStream_t stream) {
    (void)in_sizes; (void)n_in; (void)out_size; (void)ws_size;
    const float* people  = (const float*)d_in[0];
    const float* objects = (const float*)d_in[1];
    const float* context = (const float*)d_in[2];
    const float* spatial = (const float*)d_in[3];
    const float* Wvis_w  = (const float*)d_in[4];
    const float* Wvis_b  = (const float*)d_in[5];
    const float* Wspat_w = (const float*)d_in[6];
    const float* Wspat_b = (const float*)d_in[7];
    const float* Wip_w   = (const float*)d_in[8];
    const float* Wip_b   = (const float*)d_in[9];
    const float* Wref_w  = (const float*)d_in[10];
    const float* Wref_b  = (const float*)d_in[11];
    const float* Watt_w  = (const float*)d_in[12];
    const float* Watt_b  = (const float*)d_in[13];
    const float* Woh_w   = (const float*)d_in[14];
    const float* Woh_b   = (const float*)d_in[15];
    const float* Who_w   = (const float*)d_in[16];
    const float* Who_b   = (const float*)d_in[17];
    const float* Wg_w    = (const float*)d_in[18];
    const float* Wg_b    = (const float*)d_in[19];
    float* out = (float*)d_out;

    // ---- carve workspace ----
    char* ws = (char*)d_ws;
    size_t off = 0;
    auto alloc = [&](size_t bytes) -> void* {
        void* p = ws + off;
        off = (off + bytes + 255) & ~(size_t)255;
        return p;
    };
    h16*  people_h   = (h16*)alloc((size_t)BP * DD * 2);
    h16*  objects_h  = (h16*)alloc((size_t)BL * DD * 2);
    h16*  context_h  = (h16*)alloc((size_t)BB * DD * 2);
    h16*  spatial_h  = (h16*)alloc((size_t)CC * SP * 2);
    h16*  BTvis      = (h16*)alloc((size_t)PROJ * 3 * DD * 2);   // [512][3072]
    h16*  BTspat     = (h16*)alloc((size_t)PROJ * SP * 2);       // [512][32]
    h16*  BTrefip    = (h16*)alloc((size_t)32 * PROJ * 2);       // [32][512]
    h16*  BTatt      = (h16*)alloc((size_t)32 * PROJ * 2);       // [32][512]
    h16*  BToh       = (h16*)alloc((size_t)DD * DD * 2);         // [1024][1024]
    h16*  BTho       = (h16*)alloc((size_t)DD * DD * 2);         // [1024][1024]
    h16*  BTg        = (h16*)alloc((size_t)32 * 2 * DD * 2);     // [32][2048]
    h16*  a_ho_h     = (h16*)alloc((size_t)CC * PROJ * 2);
    h16*  f_ref_h    = (h16*)alloc((size_t)CC * PROJ * 2);
    h16*  people_r_h = (h16*)alloc((size_t)BP * DD * 2);
    h16*  obj_full_h = (h16*)alloc((size_t)BL * DD * 2);
    float* vis_p     = (float*)alloc((size_t)BP * PROJ * 4);
    float* vis_o     = (float*)alloc((size_t)BL * PROJ * 4);
    float* vis_c     = (float*)alloc((size_t)BB * PROJ * 4);
    float* oh_all    = (float*)alloc((size_t)BL * DD * 4);
    float* ho_all    = (float*)alloc((size_t)BP * DD * 4);
    float* adj       = (float*)alloc((size_t)BB * PP * 15 * 4);
    float* p_ref_pad = (float*)alloc((size_t)CC * 32 * 4);
    float* p_att_pad = (float*)alloc((size_t)CC * 32 * 4);
    float* g_p       = (float*)alloc((size_t)BP * 32 * 4);
    float* g_o       = (float*)alloc((size_t)BL * 32 * 4);

    auto cvt = [&](const float* x, h16* y, int n) {
        cvt_f2h<<<(n + 255) / 256, 256, 0, stream>>>(x, y, n);
    };
    // wide tiles: 32x64 per wave (MT=2, NT=4); narrow heads: 32x32 (MT=2, NT=2)
    auto gemmW = [&](const h16* A, int lda, const h16* BT, int ldb, const float* bias,
                     float* Cf, h16* Ch, int ldc, int M, int N, int K, int relu) {
        dim3 g(N / 64, M / 32);
        gemm_wmma_direct<2, 4><<<g, dim3(32), 0, stream>>>(A, lda, BT, ldb, bias,
                                                           Cf, Ch, ldc, K, relu);
    };
    auto gemmN = [&](const h16* A, int lda, const h16* BT, int ldb, const float* bias,
                     float* Cf, h16* Ch, int ldc, int M, int N, int K, int relu) {
        dim3 g(N / 32, M / 32);
        gemm_wmma_direct<2, 2><<<g, dim3(32), 0, stream>>>(A, lda, BT, ldb, bias,
                                                           Cf, Ch, ldc, K, relu);
    };

    // ---- 1) precision conversion of activations; weights -> transposed f16 ----
    cvt(people,  people_h,  BP * DD);
    cvt(objects, objects_h, BL * DD);
    cvt(context, context_h, BB * DD);
    cvt(spatial, spatial_h, CC * SP);
    transpose_f2h<<<(PROJ * 3 * DD + 255) / 256, 256, 0, stream>>>(Wvis_w, BTvis, 3 * DD, PROJ);
    transpose_f2h<<<(PROJ * SP + 255) / 256, 256, 0, stream>>>(Wspat_w, BTspat, SP, PROJ);
    transpose_f2h<<<(DD * DD + 255) / 256, 256, 0, stream>>>(Woh_w, BToh, DD, DD);
    transpose_f2h<<<(DD * DD + 255) / 256, 256, 0, stream>>>(Who_w, BTho, DD, DD);
    pad29T_f2h<<<(32 * PROJ + 255) / 256, 256, 0, stream>>>(Wref_w, Wip_w, BTrefip, PROJ);
    pad29T_f2h<<<(32 * PROJ + 255) / 256, 256, 0, stream>>>(Watt_w, nullptr, BTatt, PROJ);
    pad29T_f2h<<<(32 * 2 * DD + 255) / 256, 256, 0, stream>>>(Wg_w, nullptr, BTg, 2 * DD);

    // ---- 2) factored visual projection (concat-GEMM decomposition) ----
    // BTvis rows have length 3072; k-segments select Wvis row blocks.
    gemmW(people_h,  DD, BTvis,            3 * DD, nullptr, vis_p, nullptr, PROJ, BP, PROJ, DD, 0);
    gemmW(objects_h, DD, BTvis + DD,       3 * DD, nullptr, vis_o, nullptr, PROJ, BL, PROJ, DD, 0);
    gemmW(context_h, DD, BTvis + 2 * DD,   3 * DD, nullptr, vis_c, nullptr, PROJ, BB, PROJ, DD, 0);

    // ---- 3) spatial attention head: a_ho = relu(spatial @ Wspat + b) -> f16 ----
    gemmW(spatial_h, SP, BTspat, SP, Wspat_b, nullptr, a_ho_h, PROJ, CC, PROJ, SP, 1);

    // ---- 4) f_ref = relu(pairs@Wvis + b) * a_ho (broadcast-add fusion) ----
    fuse_fref<<<(CC * PROJ + 255) / 256, 256, 0, stream>>>(vis_p, vis_o, vis_c, Wvis_b,
                                                           a_ho_h, f_ref_h);

    // ---- 5) heads: p_ref (+ fused Wip column 29) and p_att, padded to 32 cols ----
    gemmN(f_ref_h, PROJ, BTrefip, PROJ, nullptr, p_ref_pad, nullptr, 32, CC, 32, PROJ, 0);
    gemmN(a_ho_h,  PROJ, BTatt,   PROJ, nullptr, p_att_pad, nullptr, 32, CC, 32, PROJ, 0);

    // ---- 6) graph transforms oh/ho (computed over all 16 slots; slot0 unused) ----
    gemmW(objects_h, DD, BToh, DD, Woh_b, oh_all, nullptr, DD, BL, DD, DD, 1);
    gemmW(people_h,  DD, BTho, DD, Who_b, ho_all, nullptr, DD, BP, DD, DD, 1);

    // ---- 7) message passing with sigmoid adjacency ----
    make_adj<<<(BB * PP * 15 + 255) / 256, 256, 0, stream>>>(p_ref_pad, Wip_b, adj);
    refine_people<<<(BP * DD + 255) / 256, 256, 0, stream>>>(people, adj, oh_all, people_r_h);
    refine_objects<<<(BL * DD + 255) / 256, 256, 0, stream>>>(objects, adj, ho_all, obj_full_h);

    // ---- 8) factored graph head: g_p + g_o broadcast instead of [C,2048] GEMM ----
    gemmN(people_r_h, DD, BTg,      2 * DD, nullptr, g_p, nullptr, 32, BP, 32, DD, 0);
    gemmN(obj_full_h, DD, BTg + DD, 2 * DD, nullptr, g_o, nullptr, 32, BL, 32, DD, 0);

    // ---- 9) assemble outputs (biases applied here) ----
    pack_out<<<(CC + 255) / 256, 256, 0, stream>>>(p_ref_pad, p_att_pad, g_p, g_o,
                                                   Wip_b, Wref_b, Watt_b, Wg_b, out);
}
